// NoisyTopKRouter_9517647528395
// MI455X (gfx1250) — matrix operand order repaired and checked by reading, not compiled, not run on hardware
//
#include <hip/hip_runtime.h>

typedef __attribute__((ext_vector_type(16))) __bf16 v16bf;
typedef __attribute__((ext_vector_type(8)))  float  v8f;
typedef __attribute__((ext_vector_type(4)))  float  f32x4;

#define WMMA_BF16(A, B, C) \
  __builtin_amdgcn_wmma_f32_16x16x32_bf16(false, (A), false, (B), (short)0, (C), false, false)

__device__ __forceinline__ float softplus_f(float z) {
  // stable: max(z,0) + log1p(exp(-|z|))
  return fmaxf(z, 0.0f) + log1pf(__expf(-fabsf(z)));
}

// Split an f32 into bf16 hi + bf16 residual lo (3-term product recovers
// ~fp32 GEMM accuracy with bf16 WMMA; affordable because we're memory-bound).
__device__ __forceinline__ void bf16_split(float f, __bf16& hi, __bf16& lo) {
  __bf16 h = (__bf16)f;
  hi = h;
  lo = (__bf16)(f - (float)h);
}

// ---------------------------------------------------------------------------
// Kernel 1: one-shot split of both weight matrices into bf16 hi/lo planes.
// 2 x 64x1024 f32 in, 4 x 64K bf16 out (512 KB scratch). Bandwidth-trivial.
// ---------------------------------------------------------------------------
__global__ __launch_bounds__(256)
void split_weights_kernel(const float* __restrict__ rw, const float* __restrict__ nw,
                          __bf16* __restrict__ rwHi, __bf16* __restrict__ rwLo,
                          __bf16* __restrict__ nwHi, __bf16* __restrict__ nwLo) {
  const int i = blockIdx.x * blockDim.x + threadIdx.x;   // 0 .. 65535
  __bf16 h, l;
  bf16_split(rw[i], h, l); rwHi[i] = h; rwLo[i] = l;
  bf16_split(nw[i], h, l); nwHi[i] = h; nwLo[i] = l;
}

// ---------------------------------------------------------------------------
// Kernel 2: fused dual-GEMM + softplus-noise + top-2 + 2-way softmax.
// Block = 256 threads (8 wave32), covers 128 rows x 64 experts.
// Wave w owns rows [w*16, w*16+16): 4 expert tiles x {route,noise} = 8 accs.
// ---------------------------------------------------------------------------
__global__ __launch_bounds__(256)
void noisy_topk_router_kernel(const float* __restrict__ x,
                              const __bf16* __restrict__ rwHi,
                              const __bf16* __restrict__ rwLo,
                              const __bf16* __restrict__ nwHi,
                              const __bf16* __restrict__ nwLo,
                              const float* __restrict__ route_b,
                              const float* __restrict__ noise_b,
                              const float* __restrict__ eps,
                              float* __restrict__ probs,
                              int*   __restrict__ topidx) {
  constexpr int D = 1024;
  constexpr int E = 64;

  const int tid  = threadIdx.x;
  const int wave = tid >> 5;        // 0..7
  const int lane = tid & 31;
  const int half = lane >> 4;       // 0/1
  const int l16  = lane & 15;
  const int row0 = blockIdx.x * 128 + wave * 16;   // this wave's 16 rows

  // A 16x32 bf16 fragment: lane half=0 row=l16 holds K 0..15;
  // half=1 holds K 16..31  (matches B's contiguous order)
  const float* pA = x + (size_t)(row0 + l16) * D + half * 16;

  v8f accR[4] = {};
  v8f accN[4] = {};

  for (int k0 = 0; k0 < D; k0 += 32) {
    __builtin_prefetch((const void*)(pA + k0 + 512), 0, 0);

    // ---- load + split A fragment (only conversion work in the loop) ----
    f32x4 qa[4];
    qa[0] = ((const f32x4*)(pA + k0))[0];
    qa[1] = ((const f32x4*)(pA + k0))[1];
    qa[2] = ((const f32x4*)(pA + k0))[2];
    qa[3] = ((const f32x4*)(pA + k0))[3];
    v16bf aHi, aLo;
#pragma unroll
    for (int i = 0; i < 16; ++i) {
      __bf16 h, l;
      bf16_split(qa[i >> 2][i & 3], h, l);
      aHi[i] = h; aLo[i] = l;
    }

    // ---- 4 expert tiles: B fragments are preconverted bf16, direct loads ----
#pragma unroll
    for (int t = 0; t < 4; ++t) {
      // B 32x16: lane half=0 col=l16 holds K 0..15; half=1 holds K 16..31
      const size_t o = (size_t)(t * 16 + l16) * D + half * 16 + k0;  // 32B aligned
      const v16bf brHi = *(const v16bf*)(rwHi + o);
      const v16bf brLo = *(const v16bf*)(rwLo + o);
      const v16bf bnHi = *(const v16bf*)(nwHi + o);
      const v16bf bnLo = *(const v16bf*)(nwLo + o);

      accR[t] = WMMA_BF16(aHi, brHi, accR[t]);
      accR[t] = WMMA_BF16(aLo, brHi, accR[t]);
      accR[t] = WMMA_BF16(aHi, brLo, accR[t]);
      accN[t] = WMMA_BF16(aHi, bnHi, accN[t]);
      accN[t] = WMMA_BF16(aLo, bnHi, accN[t]);
      accN[t] = WMMA_BF16(aHi, bnLo, accN[t]);
    }
  }

  // ---- epilogue: bias + softplus noise + eps scale -> wave-private LDS slab ----
  __shared__ float s_noisy[8][16][65];   // 65 pad vs 64 LDS banks
  __shared__ float s_p[8][16][2];
  __shared__ int   s_idx[8][16][2];

#pragma unroll
  for (int t = 0; t < 4; ++t) {
    const int e = t * 16 + l16;
    const float rbias = route_b[e];
    const float nbias = noise_b[e];
    // C/D layout: VGPR j -> local row = half*8 + j, col = l16
    const float* pe = eps + (size_t)(row0 + half * 8) * E + e;
#pragma unroll
    for (int j = 0; j < 8; ++j) {
      const float nscale = softplus_f(accN[t][j] + nbias);
      s_noisy[wave][half * 8 + j][e] = accR[t][j] + rbias + pe[(size_t)j * E] * nscale;
    }
  }
  __syncthreads();

  // ---- top-2 + 2-way softmax: one lane per row (wave-local data) ----
  if (lane < 16) {
    const int r = lane;
    float best = -__builtin_inff(); int bi = 0;
    for (int e = 0; e < E; ++e) {
      const float v = s_noisy[wave][r][e];
      if (v > best) { best = v; bi = e; }     // strict >: lowest index wins ties
    }
    float sec = -__builtin_inff(); int si = 0;
    for (int e = 0; e < E; ++e) {
      if (e == bi) continue;
      const float v = s_noisy[wave][r][e];
      if (v > sec) { sec = v; si = e; }
    }
    const float e1  = __expf(sec - best);
    const float inv = 1.0f / (1.0f + e1);
    s_p[wave][r][0] = inv;       s_p[wave][r][1] = e1 * inv;
    s_idx[wave][r][0] = bi;      s_idx[wave][r][1] = si;
  }
  __syncthreads();

  // ---- stream probs: each lane writes half a row (32 consecutive floats) ----
  {
    const int r  = lane >> 1;            // 0..15
    const int c0 = (lane & 1) * 32;      // 0 or 32
    const int i0 = s_idx[wave][r][0], i1 = s_idx[wave][r][1];
    const float p0 = s_p[wave][r][0], p1 = s_p[wave][r][1];
    float* dst = probs + (size_t)(row0 + r) * E + c0;
#pragma unroll
    for (int c = 0; c < 32; ++c) {
      const int e = c0 + c;
      dst[c] = (e == i0) ? p0 : ((e == i1) ? p1 : 0.0f);
    }
  }
  if (lane < 16) {
    topidx[(size_t)(row0 + lane) * 2 + 0] = s_idx[wave][lane][0];
    topidx[(size_t)(row0 + lane) * 2 + 1] = s_idx[wave][lane][1];
  }
}

extern "C" void kernel_launch(void* const* d_in, const int* in_sizes, int n_in,
                              void* d_out, int out_size, void* d_ws, size_t ws_size,
                              hipStream_t stream) {
  const float* x       = (const float*)d_in[0];
  const float* route_w = (const float*)d_in[1];
  const float* route_b = (const float*)d_in[2];
  const float* noise_w = (const float*)d_in[3];
  const float* noise_b = (const float*)d_in[4];
  const float* eps     = (const float*)d_in[5];

  const int D = 1024;
  const int E = 64;
  const int N = in_sizes[0] / D;           // 16384
  const int W = E * D;                     // 65536 weights per matrix

  float* probs  = (float*)d_out;                       // [N,64] f32
  int*   topidx = (int*)(probs + (size_t)N * E);       // [N,2] i32 after probs

  // scratch: 4 bf16 planes of 64K elements = 512 KB
  __bf16* rwHi = (__bf16*)d_ws;
  __bf16* rwLo = rwHi + W;
  __bf16* nwHi = rwLo + W;
  __bf16* nwLo = nwHi + W;

  split_weights_kernel<<<dim3(W / 256), dim3(256), 0, stream>>>(
      route_w, noise_w, rwHi, rwLo, nwHi, nwLo);

  noisy_topk_router_kernel<<<dim3(N / 128), dim3(256), 0, stream>>>(
      x, rwHi, rwLo, nwHi, nwLo, route_b, noise_b, eps, probs, topidx);
}
